// Conv2d_84808424226962
// MI455X (gfx1250) — compile-verified, matrix operand order checked by reading, not run on hardware
//
#include <hip/hip_runtime.h>

typedef __attribute__((ext_vector_type(16))) __bf16        v16bf;
typedef __attribute__((ext_vector_type(8)))  float          v8f;
typedef __attribute__((ext_vector_type(8)))  unsigned short v8us;
typedef __attribute__((ext_vector_type(16))) unsigned short v16us;

#define C_IN   256
#define K_OUT  256
#define Hdim   56
#define HW2    3136               // 56*56
#define Hp     58                 // padded spatial dim
#define HWp    (Hp * Hp)          // 3364
#define NBATCH 32
#define M_TOT  (NBATCH * HW2)     // 100352 output pixels
#define WS_WT_ELEMS (9 * K_OUT * C_IN)            // 589824 bf16
#define WS_IN_ELEMS (NBATCH * HWp * C_IN)         // 27557888 bf16 (padded NHWC)

__device__ __forceinline__ unsigned short f32_to_bf16_rne(float f) {
  unsigned int u = __float_as_uint(f);
  u += 0x7FFFu + ((u >> 16) & 1u);   // round-to-nearest-even
  return (unsigned short)(u >> 16);
}

__device__ __forceinline__ v16bf make_frag(v8us lo, v8us hi) {
  v16us u = __builtin_shufflevector(lo, hi, 0,1,2,3,4,5,6,7,8,9,10,11,12,13,14,15);
  return __builtin_bit_cast(v16bf, u);
}

// ---------------- Prologue 0: zero-fill padded input region (16B/thread) -----------
__global__ __launch_bounds__(256) void zero_fill16(uint4* __restrict__ p) {
  p[blockIdx.x * 256 + threadIdx.x] = (uint4){0u, 0u, 0u, 0u};
}

// ---------------- Prologue 1: weights fp32 OIHW -> bf16 [rs][ko][c] ----------------
__global__ __launch_bounds__(256) void pack_weights(const float* __restrict__ wt,
                                                    unsigned short* __restrict__ wp) {
  int idx = blockIdx.x * 256 + threadIdx.x;   // (rs*256 + ko)*256 + c
  int c  = idx & 255;
  int ko = (idx >> 8) & 255;
  int rs = idx >> 16;
  wp[idx] = f32_to_bf16_rne(wt[(ko * 256 + c) * 9 + rs]);
}

// ------- Prologue 2: input fp32 NCHW -> bf16 padded NHWC [n][58][58][256] ----------
__global__ __launch_bounds__(256) void nchw_to_nhwc_pad_bf16(
    const float* __restrict__ ip, unsigned short* __restrict__ op) {
  __shared__ unsigned short tile[64][72];     // [c][hw], padded stride
  const int n   = blockIdx.z;
  const int hw0 = blockIdx.x * 64;            // 3136 = 49*64, exact
  const int c0  = blockIdx.y * 64;
  const int tx  = threadIdx.x & 63;
  const int ty  = threadIdx.x >> 6;

  const float* inBase = ip + ((size_t)n * C_IN + c0) * HW2 + hw0;
#pragma unroll
  for (int j = 0; j < 16; ++j) {
    const int c = ty + j * 4;
    tile[c][tx] = f32_to_bf16_rne(inBase[(size_t)c * HW2 + tx]);  // coalesced reads
  }
  __syncthreads();
  unsigned short* outN = op + (size_t)n * HWp * C_IN;
#pragma unroll
  for (int j = 0; j < 16; ++j) {
    const int ghw = hw0 + ty + j * 4;
    const int h   = ghw / Hdim;
    const int w   = ghw - h * Hdim;
    // interior pixel (h,w) -> padded coords (h+1, w+1); lanes along c: coalesced
    outN[(size_t)((h + 1) * Hp + (w + 1)) * C_IN + c0 + tx] = tile[tx][ty + j * 4];
  }
}

// ---------------- Main: implicit-GEMM conv, no LDS, no predication -----------------
// Block = 128 kout x 128 pixels, 8 waves; wave = 2 kt x 4 pt = 8 accs, 8 WMMA/k-step.
// Addressing: loop-invariant unsigned per-lane BYTE offsets + uniform per-rs base
// pointers + compile-time immediates -> global_load_b128 saddr+voffset form.
__global__ __launch_bounds__(256) void conv3x3_wmma_main(
    const unsigned short* __restrict__ wp,    // bf16 [rs][ko][c]
    const unsigned short* __restrict__ inb,   // bf16 padded NHWC [n][58][58][256]
    const float* __restrict__ bias,
    float* __restrict__ out)                  // fp32 NCHW
{
  const int tid  = threadIdx.x;
  const int wave = tid >> 5;
  const int lane = tid & 31;
  const int half = lane >> 4;
  const int l15  = lane & 15;
  const int kt0  = (wave & 3) * 2;            // 2 kout subtiles
  const int pt0  = (wave >> 2) * 4;           // 4 pixel subtiles
  const int pix0 = blockIdx.x * 128;
  const int ko0  = blockIdx.y * 128;

  const unsigned char* wpB  = (const unsigned char*)wp;
  const unsigned char* inbB = (const unsigned char*)inb;

  // Per-lane BYTE offsets, invariant over the whole main loop
  unsigned pixOffB[4];
#pragma unroll
  for (int p = 0; p < 4; ++p) {
    const int m  = pix0 + (pt0 + p) * 16 + l15;
    const int n  = m / HW2;
    const int hw = m - n * HW2;
    const int h  = hw / Hdim;
    const int w  = hw - h * Hdim;
    pixOffB[p] = (unsigned)(((n * HWp + h * Hp + w) * C_IN + half * 8) * 2);
  }
  unsigned aOffB[2];
#pragma unroll
  for (int t = 0; t < 2; ++t)
    aOffB[t] = (unsigned)((((ko0 + (kt0 + t) * 16 + l15) * C_IN) + half * 8) * 2);

  v8f acc[2][4];
#pragma unroll
  for (int t = 0; t < 2; ++t)
#pragma unroll
    for (int p = 0; p < 4; ++p)
      acc[t][p] = (v8f){0.f,0.f,0.f,0.f,0.f,0.f,0.f,0.f};

  for (int rs = 0; rs < 9; ++rs) {
    const int r = rs / 3, s = rs - r * 3;
    // uniform (SGPR) base pointers for this filter tap
    const unsigned char* aBase = wpB  + (size_t)(rs * K_OUT * C_IN * 2);
    const unsigned char* bBase = inbB + (size_t)((r * Hp + s) * C_IN * 2);

#pragma unroll
    for (int c0 = 0; c0 < C_IN; c0 += 32) {
      v16bf aF[2];
#pragma unroll
      for (int t = 0; t < 2; ++t) {
        const v8us* pa = (const v8us*)(aBase + aOffB[t] + c0 * 2);
        aF[t] = make_frag(pa[0], pa[2]);         // chunks at imm +0 / +32 bytes
      }
      v16bf bF[4];
#pragma unroll
      for (int p = 0; p < 4; ++p) {
        const v8us* pb = (const v8us*)(bBase + pixOffB[p] + c0 * 2);
        bF[p] = make_frag(pb[0], pb[2]);
      }
#pragma unroll
      for (int t = 0; t < 2; ++t)
#pragma unroll
        for (int p = 0; p < 4; ++p)
          acc[t][p] = __builtin_amdgcn_wmma_f32_16x16x32_bf16(
              false, aF[t], false, bF[p], (short)0, acc[t][p], false, false);
    }
  }

  // Epilogue: lane = pixel (coalesced hw), VGPR v -> kout = v + 8*half
#pragma unroll
  for (int t = 0; t < 2; ++t) {
#pragma unroll
    for (int p = 0; p < 4; ++p) {
      const int mp = pix0 + (pt0 + p) * 16 + l15;
      const int n  = mp / HW2;
      const int hw = mp - n * HW2;
#pragma unroll
      for (int v = 0; v < 8; ++v) {
        const int ko = ko0 + (kt0 + t) * 16 + v + 8 * half;
        out[(size_t)(n * K_OUT + ko) * HW2 + hw] = acc[t][p][v] + bias[ko];
      }
    }
  }
}

extern "C" void kernel_launch(void* const* d_in, const int* in_sizes, int n_in,
                              void* d_out, int out_size, void* d_ws, size_t ws_size,
                              hipStream_t stream) {
  const float* ip   = (const float*)d_in[0];
  const float* wt   = (const float*)d_in[1];
  const float* bias = (const float*)d_in[2];
  float* out = (float*)d_out;

  unsigned short* wsu = (unsigned short*)d_ws;
  unsigned short* wp  = wsu;                 // 1.18 MB packed bf16 weights
  unsigned short* inb = wsu + WS_WT_ELEMS;   // 55.1 MB padded NHWC bf16 input

  // zero-fill padded input region: 55,115,776 B / 16 = 3,444,736 uint4 (exact /256)
  zero_fill16<<<(WS_IN_ELEMS * 2 / 16) / 256, 256, 0, stream>>>((uint4*)inb);
  pack_weights<<<WS_WT_ELEMS / 256, 256, 0, stream>>>(wt, wp);
  nchw_to_nhwc_pad_bf16<<<dim3(HW2 / 64, C_IN / 64, NBATCH), 256, 0, stream>>>(ip, inb);
  conv3x3_wmma_main<<<dim3(M_TOT / 128, K_OUT / 128), 256, 0, stream>>>(wp, inb, bias, out);
}
